// Attention_36172214567347
// MI455X (gfx1250) — matrix operand order repaired and checked by reading, not compiled, not run on hardware
//
#include <hip/hip_runtime.h>

#define B_ 32
#define S_ 2048
#define D_ 512
#define A_ 128

typedef __attribute__((ext_vector_type(16))) __bf16 v16bf;
typedef __attribute__((ext_vector_type(8)))  __bf16 v8bf;
typedef __attribute__((ext_vector_type(8)))  float  v8f;
typedef __attribute__((ext_vector_type(4)))  unsigned int v4u;
typedef __attribute__((ext_vector_type(8)))  int v8i;
typedef __attribute__((ext_vector_type(4)))  int v4i;

#define HAVE_TDM (__has_builtin(__builtin_amdgcn_tensor_load_to_lds) && \
                  __has_builtin(__builtin_amdgcn_s_wait_tensorcnt))

// ---------------- phase 1a: convert mW/cW fp32 -> bf16 (workspace) --------
__global__ void convert_weights_kernel(const float* __restrict__ mW,
                                       const float* __restrict__ cW,
                                       __bf16* __restrict__ wm,
                                       __bf16* __restrict__ wc) {
  int idx = blockIdx.x * blockDim.x + threadIdx.x;
  if (idx < A_ * D_) {
    wm[idx] = (__bf16)mW[idx];
    wc[idx] = (__bf16)cW[idx];
  }
}

// ---------------- phase 1b: weight-normed output vectors ------------------
__global__ void weff_kernel(const float* __restrict__ mvv, const float* __restrict__ mvg,
                            const float* __restrict__ cvv, const float* __restrict__ cvg,
                            float* __restrict__ weffm, float* __restrict__ weffc) {
  __shared__ float red[A_];
  int t = threadIdx.x;
  float vm = mvv[t];
  red[t] = vm * vm;
  __syncthreads();
  for (int off = 64; off > 0; off >>= 1) { if (t < off) red[t] += red[t + off]; __syncthreads(); }
  float nm = sqrtf(red[0]);
  __syncthreads();
  weffm[t] = mvg[0] * vm / nm;
  float vc = cvv[t];
  red[t] = vc * vc;
  __syncthreads();
  for (int off = 64; off > 0; off >>= 1) { if (t < off) red[t] += red[t + off]; __syncthreads(); }
  float nc = sqrtf(red[0]);
  weffc[t] = cvg[0] * vc / nc;
}

// ---------------- phase 1c: dec @ V^T + bias (per batch, per head) --------
__global__ void decterm_kernel(const float* __restrict__ dec,
                               const float* __restrict__ mV, const float* __restrict__ mb,
                               const float* __restrict__ cV, const float* __restrict__ cb,
                               float* __restrict__ decm, float* __restrict__ decc) {
  int b = blockIdx.x >> 1;
  int which = blockIdx.x & 1;
  int a = threadIdx.x;
  const float* V    = which ? cV : mV;
  const float* bias = which ? cb : mb;
  float* out        = which ? decc : decm;
  const float* dv = dec + (size_t)b * D_;
  const float* vr = V + (size_t)a * D_;
  float acc = bias[a];
  for (int d = 0; d < D_; ++d) acc += dv[d] * vr[d];
  out[b * A_ + a] = acc;
}

// -------- TDM helper: DMA a contiguous 16KB slab (4096 dwords) to LDS -----
__device__ __forceinline__ void tdm_copy_16k_to_lds(const void* gptr, unsigned int lds_byte_addr) {
#if HAVE_TDM
  unsigned long long ga = (unsigned long long)(uintptr_t)gptr;
  v4u g0;
  g0[0] = 1u;                                   // count=1, gather off
  g0[1] = lds_byte_addr;                        // lds_addr
  g0[2] = (unsigned int)ga;                     // global_addr[31:0]
  g0[3] = (unsigned int)((ga >> 32) & 0x01FFFFFFu) | 0x80000000u; // addr[56:32] | type=2
  v8i g1;
  g1[0] = 0x00020000;                           // data_size = 4B (code 2), mask=0
  g1[1] = 0x10000000;                           // tensor_dim0 = 4096 (low 16 in [31:16])
  g1[2] = 0x00010000;                           // tensor_dim1 = 1
  g1[3] = 0x10000000;                           // tile_dim0 = 4096
  g1[4] = 0x00000001;                           // tile_dim1 = 1, tile_dim2 = 0
  g1[5] = 0x00001000;                           // tensor_dim0_stride = 4096
  g1[6] = 0;
  g1[7] = 0;
  v4i g2 = {0, 0, 0, 0};
  v4i g3 = {0, 0, 0, 0};
  v8i g4 = {0, 0, 0, 0, 0, 0, 0, 0};            // unused trailing group (6-arg form)
  __builtin_amdgcn_tensor_load_to_lds(g0, g1, g2, g3, g4, 0);
#else
  (void)gptr; (void)lds_byte_addr;
#endif
}

__device__ __forceinline__ unsigned int lds_offset_of(const void* p) {
#if defined(__HIP_DEVICE_COMPILE__)
  return (unsigned int)(uintptr_t)(__attribute__((address_space(3))) const void*)p;
#else
  (void)p; return 0;
#endif
}

// ---------------- phase 2: fused dual-energy GEMM via WMMA bf16 -----------
// One block = 2 waves; 32x512 bf16 A-tile in LDS; per-slab weights staged
// into LDS by the Tensor Data Mover (fallback: cooperative copy).
__global__ __launch_bounds__(64) void energy_wmma_kernel(
    const float* __restrict__ enc,
    const __bf16* __restrict__ wm, const __bf16* __restrict__ wc,
    const float* __restrict__ decm, const float* __restrict__ decc,
    const float* __restrict__ weffm, const float* __restrict__ weffc,
    const float* __restrict__ mvb, const float* __restrict__ mr,
    const float* __restrict__ cvb, const float* __restrict__ cr,
    float* __restrict__ mono_e, float* __restrict__ u_out) {
  const int ROWS = 32;
  __shared__ __bf16 Atile[ROWS * D_];   // 32 KB
  __shared__ __bf16 BsM[16 * D_];       // 16 KB: current mono weight slab
  __shared__ __bf16 BsC[16 * D_];       // 16 KB: current chunk weight slab
  int b    = blockIdx.x / (S_ / ROWS);
  int row0 = (blockIdx.x % (S_ / ROWS)) * ROWS;
  int tid  = threadIdx.x;

  // cooperative fp32 -> bf16 staging of the 32x512 encoder tile
  const float* encblk = enc + ((size_t)b * S_ + row0) * D_;
  for (int i = tid; i < ROWS * D_ / 4; i += blockDim.x) {
    float4 v = ((const float4*)encblk)[i];
    int e = i * 4;
    Atile[e + 0] = (__bf16)v.x; Atile[e + 1] = (__bf16)v.y;
    Atile[e + 2] = (__bf16)v.z; Atile[e + 3] = (__bf16)v.w;
  }

  int wave = tid >> 5;
  int lane = tid & 31;
  int lrow = lane & 15;
  int half = lane >> 4;            // 0: lanes 0-15, 1: lanes 16-31
  int abase = half ? 16 : 0;       // A-fragment K sub-offset (contiguous K, matches B)
  int bbase = half ? 16 : 0;       // B-fragment K sub-offset (contiguous K, matches A)
  const __bf16* aRow = &Atile[(wave * 16 + lrow) * D_];

  float rs_m[8], rs_c[8];
#pragma unroll
  for (int r = 0; r < 8; ++r) { rs_m[r] = 0.f; rs_c[r] = 0.f; }

  for (int nt = 0; nt < A_ / 16; ++nt) {
    // ---- stage this slab's weights (16 cols x 512 K, contiguous 16KB) ----
#if HAVE_TDM
    if (tid < 32) {
      tdm_copy_16k_to_lds(wm + (size_t)nt * 16 * D_, lds_offset_of(BsM));
      tdm_copy_16k_to_lds(wc + (size_t)nt * 16 * D_, lds_offset_of(BsC));
      __builtin_amdgcn_s_wait_tensorcnt(0);
    }
#else
    {
      const uint4* gm = (const uint4*)(wm + (size_t)nt * 16 * D_);
      const uint4* gc = (const uint4*)(wc + (size_t)nt * 16 * D_);
      uint4* sm = (uint4*)BsM;
      uint4* sc = (uint4*)BsC;
      for (int i = tid; i < 16 * D_ * 2 / 16; i += blockDim.x) {
        sm[i] = gm[i];
        sc[i] = gc[i];
      }
    }
#endif
    __syncthreads();

    int acol = nt * 16 + lrow;     // output column (attention dim)
    const __bf16* bm_row = BsM + lrow * D_ + bbase;
    const __bf16* bc_row = BsC + lrow * D_ + bbase;
    v8f cm = {}; v8f cc = {};
#pragma unroll 4
    for (int kk = 0; kk < D_; kk += 32) {
      v16bf a = *(const v16bf*)(aRow + kk + abase);
      v16bf bm = *(const v16bf*)(bm_row + kk);
      v16bf bc = *(const v16bf*)(bc_row + kk);
      cm = __builtin_amdgcn_wmma_f32_16x16x32_bf16(false, a, false, bm, (short)0, cm, false, false);
      cc = __builtin_amdgcn_wmma_f32_16x16x32_bf16(false, a, false, bc, (short)0, cc, false, false);
    }
    // epilogue: tanh + weighted reduction over this 16-column slab
    float addm = decm[b * A_ + acol];
    float addc = decc[b * A_ + acol];
    float wem  = weffm[acol];
    float wec  = weffc[acol];
#pragma unroll
    for (int r = 0; r < 8; ++r) {
      rs_m[r] += tanhf(cm[r] + addm) * wem;
      rs_c[r] += tanhf(cc[r] + addc) * wec;
    }
    __syncthreads();   // slab consumed; safe to overwrite next iteration
  }

  // butterfly-reduce the 16 column-partials inside each half-wave
#pragma unroll
  for (int r = 0; r < 8; ++r) {
    float m = rs_m[r], c = rs_c[r];
#pragma unroll
    for (int msk = 8; msk >= 1; msk >>= 1) {
      m += __shfl_xor(m, msk, 32);
      c += __shfl_xor(c, msk, 32);
    }
    rs_m[r] = m; rs_c[r] = c;
  }
  if (lrow == 0) {
    float sm = mvb[0] + mr[0];
    float sc = cvb[0] + cr[0];
    int mrow0 = row0 + wave * 16 + (half ? 8 : 0);   // C layout: VGPR r -> row r (+8 upper half)
#pragma unroll
    for (int r = 0; r < 8; ++r) {
      mono_e[(size_t)b * S_ + mrow0 + r] = rs_m[r] + sm;
      u_out [(size_t)b * S_ + mrow0 + r] = rs_c[r] + sc;
    }
  }
}

// ---------------- phase 3: per-batch scans (alpha, beta) ------------------
__device__ __forceinline__ float block_scan_excl(float v, float* s, int tid) {
  __syncthreads();
  s[tid] = v;
  __syncthreads();
  for (int off = 1; off < 256; off <<= 1) {
    float mine = s[tid];
    float prev = (tid >= off) ? s[tid - off] : 0.f;
    __syncthreads();
    s[tid] = mine + prev;
    __syncthreads();
  }
  return s[tid] - v;   // exclusive prefix
}

__device__ __forceinline__ float block_max(float v, float* s, int tid) {
  __syncthreads();
  s[tid] = v;
  __syncthreads();
  for (int off = 128; off > 0; off >>= 1) {
    if (tid < off) s[tid] = fmaxf(s[tid], s[tid + off]);
    __syncthreads();
  }
  float r = s[0];
  __syncthreads();
  return r;
}

__global__ __launch_bounds__(256) void mocha_scan_kernel(
    const float* __restrict__ mono_e, const float* __restrict__ u_in,
    const float* __restrict__ noise,  const float* __restrict__ prev_alpha,
    float* __restrict__ alpha_out, float* __restrict__ beta_out,
    float* __restrict__ beta_ws) {
  __shared__ float s_scan[256];
  __shared__ float s_arr[S_];
  int b = blockIdx.x, tid = threadIdx.x;
  const int base = tid * 8;
  const size_t boff = (size_t)b * S_;

  // p_select and safe cumprod of (1 - p)
  float p[8], lp[8], cum[8];
  float lsum = 0.f;
#pragma unroll
  for (int i = 0; i < 8; ++i) {
    float e = mono_e[boff + base + i] + noise[boff + base + i];
    float pi = 1.f / (1.f + expf(-e));
    p[i] = pi;
    float x = fminf(fmaxf(1.f - pi, 1e-10f), 1.f);
    lp[i] = logf(x);
    lsum += lp[i];
  }
  float run = block_scan_excl(lsum, s_scan, tid);
#pragma unroll
  for (int i = 0; i < 8; ++i) { run += lp[i]; cum[i] = expf(run); }

  // alpha = p * cumprod * cumsum(prev_alpha / cumprod)
  float q[8]; lsum = 0.f;
#pragma unroll
  for (int i = 0; i < 8; ++i) { q[i] = prev_alpha[boff + base + i] / cum[i]; lsum += q[i]; }
  run = block_scan_excl(lsum, s_scan, tid);
  float alpha[8];
#pragma unroll
  for (int i = 0; i < 8; ++i) {
    run += q[i];
    alpha[i] = p[i] * cum[i] * run;
    alpha_out[boff + base + i] = alpha[i];
  }

  // chunk energies: stable exp
  float uu[8], eu[8];
  float um = -3.4e38f;
#pragma unroll
  for (int i = 0; i < 8; ++i) { uu[i] = u_in[boff + base + i]; um = fmaxf(um, uu[i]); }
  float gmax = block_max(um, s_scan, tid);
#pragma unroll
  for (int i = 0; i < 8; ++i) eu[i] = fmaxf(expf(uu[i] - gmax), 1e-5f);

  // denom = moving_sum(exp_u, back=7, forward=0) via inclusive cumsum
  lsum = 0.f;
#pragma unroll
  for (int i = 0; i < 8; ++i) lsum += eu[i];
  run = block_scan_excl(lsum, s_scan, tid);
  float cse[8];
#pragma unroll
  for (int i = 0; i < 8; ++i) { run += eu[i]; cse[i] = run; s_arr[base + i] = run; }
  __syncthreads();
  float g[8];
#pragma unroll
  for (int i = 0; i < 8; ++i) {
    int s = base + i;
    float lo = (s >= 8) ? s_arr[s - 8] : 0.f;
    g[i] = alpha[i] / (cse[i] - lo);
  }
  __syncthreads();

  // beta = exp_u * moving_sum(alpha/denom, back=0, forward=7)
  lsum = 0.f;
#pragma unroll
  for (int i = 0; i < 8; ++i) lsum += g[i];
  run = block_scan_excl(lsum, s_scan, tid);
#pragma unroll
  for (int i = 0; i < 8; ++i) { run += g[i]; s_arr[base + i] = run; }
  __syncthreads();
#pragma unroll
  for (int i = 0; i < 8; ++i) {
    int s = base + i;
    float hi = s_arr[min(s + 7, S_ - 1)];
    float lo = (s > 0) ? s_arr[s - 1] : 0.f;
    float beta = eu[i] * (hi - lo);
    beta_out[boff + s] = beta;
    beta_ws[boff + s]  = beta;
  }
}

// ---------------- phase 4: context = enc^T @ beta -------------------------
__global__ __launch_bounds__(128) void context_kernel(
    const float* __restrict__ enc, const float* __restrict__ beta,
    float* __restrict__ out_ctx) {
  __shared__ float sb[128];
  int b  = blockIdx.x >> 2;
  int d  = ((blockIdx.x & 3) * 128) + threadIdx.x;
  float acc = 0.f;
  for (int s0 = 0; s0 < S_; s0 += 128) {
    sb[threadIdx.x] = beta[(size_t)b * S_ + s0 + threadIdx.x];
    __syncthreads();
    const float* ep = enc + ((size_t)b * S_ + s0) * D_ + d;
    for (int j = 0; j < 128; ++j) acc += ep[(size_t)j * D_] * sb[j];
    __syncthreads();
  }
  out_ctx[(size_t)b * D_ + d] = acc;
}

// --------------------------------------------------------------------------
extern "C" void kernel_launch(void* const* d_in, const int* in_sizes, int n_in,
                              void* d_out, int out_size, void* d_ws, size_t ws_size,
                              hipStream_t stream) {
  (void)in_sizes; (void)n_in; (void)out_size; (void)ws_size;
  const float* enc   = (const float*)d_in[0];
  const float* dec   = (const float*)d_in[1];
  const float* pa    = (const float*)d_in[2];
  const float* noise = (const float*)d_in[3];
  const float* mW  = (const float*)d_in[4];
  const float* mV  = (const float*)d_in[5];
  const float* mb  = (const float*)d_in[6];
  const float* mvv = (const float*)d_in[7];
  const float* mvg = (const float*)d_in[8];
  const float* mvb = (const float*)d_in[9];
  const float* mr  = (const float*)d_in[10];
  const float* cW  = (const float*)d_in[11];
  const float* cV  = (const float*)d_in[12];
  const float* cb  = (const float*)d_in[13];
  const float* cvv = (const float*)d_in[14];
  const float* cvg = (const float*)d_in[15];
  const float* cvb = (const float*)d_in[16];
  const float* cr  = (const float*)d_in[17];

  char* ws = (char*)d_ws;
  __bf16* wm   = (__bf16*)(ws);                       // 131072 B
  __bf16* wc   = (__bf16*)(ws + 131072);              // 131072 B
  float* weffm = (float*)(ws + 262144);               // 512 B
  float* weffc = (float*)(ws + 262656);               // 512 B
  float* decm  = (float*)(ws + 263168);               // 16384 B
  float* decc  = (float*)(ws + 279552);               // 16384 B
  float* mono  = (float*)(ws + 295936);               // 262144 B
  float* uarr  = (float*)(ws + 558080);               // 262144 B
  float* betaw = (float*)(ws + 820224);               // 262144 B (total ~1.03 MB)

  float* out_ctx   = (float*)d_out;                   // [B, D]
  float* out_alpha = out_ctx + B_ * D_;               // [B, S]
  float* out_beta  = out_alpha + B_ * S_;             // [B, S]

  convert_weights_kernel<<<(A_ * D_ + 255) / 256, 256, 0, stream>>>(mW, cW, wm, wc);
  weff_kernel<<<1, A_, 0, stream>>>(mvv, mvg, cvv, cvg, weffm, weffc);
  decterm_kernel<<<B_ * 2, A_, 0, stream>>>(dec, mV, mb, cV, cb, decm, decc);
  energy_wmma_kernel<<<B_ * (S_ / 32), 64, 0, stream>>>(
      enc, wm, wc, decm, decc, weffm, weffc, mvb, mr, cvb, cr, mono, uarr);
  mocha_scan_kernel<<<B_, 256, 0, stream>>>(mono, uarr, noise, pa,
                                            out_alpha, out_beta, betaw);
  context_kernel<<<B_ * (D_ / 128), 128, 0, stream>>>(enc, betaw, out_ctx);
}
